// LSTMModel_25288767439585
// MI455X (gfx1250) — matrix operand order repaired and checked by reading, not compiled, not run on hardware
//
#include <hip/hip_runtime.h>
#include <hip/hip_bf16.h>

#define VOCAB 50257
#define DIMS 512
#define SEQ 2048
#define GDIM (4 * DIMS)                 // 2048 fused gate columns
#define NTILE ((VOCAB + 15) / 16)       // 3142 column tiles for logits
#define WTILE (GDIM / 16)               // 128 column tiles for fused gates
#define NWG_SCAN 16                     // cooperative workgroups in the scan
#define JPB (DIMS / NWG_SCAN)           // 32 hidden units per scan block
#define ROWS_PER_BLK 64                 // A rows staged per GEMM block
#define FRAG_PER_TILE (16 * 32)         // v16bf fragments per 16-col tile (kc*lane)

typedef __attribute__((ext_vector_type(16))) __bf16 v16bf;
typedef __attribute__((ext_vector_type(4)))  __bf16 v4bf;
typedef __attribute__((ext_vector_type(8)))  float  v8f;

// ---------------- WMMA A-fragment from LDS (wave32, 16x16x32 bf16) -------------
// A 16x32 (MxK): lanes 0-15 -> M, K 0..15; lanes 16-31 -> same M, K 16..31 (contiguous).
__device__ __forceinline__ v16bf load_a_lds(const __bf16* sA, int lane, int kc) {
  const int m  = lane & 15;
  const int kb = kc + ((lane >> 4) << 4);
  const __bf16* p = sA + m * DIMS;
  v16bf a;
#pragma unroll
  for (int v = 0; v < 8; ++v) {
    const int k = kb + (v << 1);
    a[2 * v]     = p[k];
    a[2 * v + 1] = p[k + 1];
  }
  return a;
}

// ---------------- generic B packer: f32 [K=512][ldb] -> bf16 fragment order ----
// Fragment f = (c*16 + kc)*32 + lane holds that lane's 16 bf16 B values.
__global__ __launch_bounds__(512) void packB_kernel(const float* __restrict__ B,
                                                    int ldb, int ncols,
                                                    __bf16* __restrict__ PB) {
  const int c    = blockIdx.x;
  const int tid  = threadIdx.x;       // 512 = 16 kc * 32 lanes
  const int kc   = tid >> 5;
  const int lane = tid & 31;
  int n = c * 16 + (lane & 15);
  if (n >= ncols) n = ncols - 1;      // duplicate last column; stores are guarded later
  const int kb = kc * 32 + ((lane >> 4) << 4);
  __bf16* dst = PB + (((size_t)(c * 16 + kc) * 32) + lane) * 16;
  const float* p = B + n;
#pragma unroll
  for (int v = 0; v < 8; ++v) {
    dst[2 * v]     = (__bf16)p[(size_t)(kb + 2 * v) * ldb];
    dst[2 * v + 1] = (__bf16)p[(size_t)(kb + 2 * v + 1) * ldb];
  }
}

// ---------------- Kernel 1: embed + fused 4-gate input GEMM (packed W) ---------
__global__ __launch_bounds__(256) void pregate_kernel(
    const int* __restrict__ idx, const float* __restrict__ emb,
    const __bf16* __restrict__ PW,
    const float* __restrict__ bf, const float* __restrict__ bi,
    const float* __restrict__ bc, const float* __restrict__ bo,
    float* __restrict__ XG) {
  __shared__ __bf16 sA[ROWS_PER_BLK * DIMS];     // 64 KB
  __shared__ int sIdx[ROWS_PER_BLK];
  const int row0 = blockIdx.x * ROWS_PER_BLK;
  const int tid  = threadIdx.x;
  if (tid < ROWS_PER_BLK) sIdx[tid] = idx[row0 + tid];
  __syncthreads();
  for (int e4 = tid; e4 < ROWS_PER_BLK * DIMS / 4; e4 += 256) {
    const int r = e4 >> 7, k4 = e4 & 127;
    const float4 f = ((const float4*)(emb + (size_t)sIdx[r] * DIMS))[k4];
    v4bf o; o[0] = (__bf16)f.x; o[1] = (__bf16)f.y; o[2] = (__bf16)f.z; o[3] = (__bf16)f.w;
    ((v4bf*)sA)[e4] = o;
  }
  __syncthreads();

  const int wave = tid >> 5, lane = tid & 31;
  const int c = blockIdx.y * 8 + wave;           // col tile 0..127 (gate = c>>5)
  const int g = c >> 5;
  const float* bias = (g == 0) ? bf : (g == 1) ? bi : (g == 2) ? bc : bo;

  const v16bf* pw = (const v16bf*)PW + (size_t)c * FRAG_PER_TILE + lane;
  v8f acc0 = {}, acc1 = {}, acc2 = {}, acc3 = {};
#pragma unroll 4
  for (int kc = 0; kc < 16; ++kc) {
    const v16bf b = pw[kc * 32];
    v16bf a0 = load_a_lds(sA + 0 * 16 * DIMS, lane, kc * 32);
    v16bf a1 = load_a_lds(sA + 1 * 16 * DIMS, lane, kc * 32);
    v16bf a2 = load_a_lds(sA + 2 * 16 * DIMS, lane, kc * 32);
    v16bf a3 = load_a_lds(sA + 3 * 16 * DIMS, lane, kc * 32);
    acc0 = __builtin_amdgcn_wmma_f32_16x16x32_bf16(false, a0, false, b, (short)0, acc0, false, false);
    acc1 = __builtin_amdgcn_wmma_f32_16x16x32_bf16(false, a1, false, b, (short)0, acc1, false, false);
    acc2 = __builtin_amdgcn_wmma_f32_16x16x32_bf16(false, a2, false, b, (short)0, acc2, false, false);
    acc3 = __builtin_amdgcn_wmma_f32_16x16x32_bf16(false, a3, false, b, (short)0, acc3, false, false);
  }

  const int n    = lane & 15;
  const float bv = bias[((c & 31) << 4) + n];
  const int mb   = (lane >> 4) << 3;
  const int colg = (c << 4) + n;
  v8f accs[4] = {acc0, acc1, acc2, acc3};
#pragma unroll
  for (int rt = 0; rt < 4; ++rt)
#pragma unroll
    for (int r = 0; r < 8; ++r)
      XG[(size_t)(row0 + rt * 16 + mb + r) * GDIM + colg] = accs[rt][r] + bv;
}

// ---------------- Kernel 2a: init scan state ----------------
__global__ void scan_init_kernel(float* __restrict__ hbuf, unsigned* __restrict__ barcnt) {
  const int i = blockIdx.x * blockDim.x + threadIdx.x;
  if (i < 2 * DIMS) hbuf[i] = 0.f;
  if (i == 0) *barcnt = 0u;
}

// ---------------- Kernel 2b: cooperative LSTM scan over 16 WGPs ----------------
__global__ __launch_bounds__(256) void lstm_scan_multi(
    const float* __restrict__ XG,
    const float* __restrict__ Uf, const float* __restrict__ Ui,
    const float* __restrict__ Uc, const float* __restrict__ Uo,
    float* __restrict__ H, float* __restrict__ hbuf, unsigned* __restrict__ barcnt) {
  __shared__ float sh[DIMS];          // staged h_{t-1}
  __shared__ float part[256];         // per-thread partial dots
  __shared__ float sg[4][JPB];        // gate activations
  __shared__ float sc[JPB];           // cell state (block-local)
  const int b    = blockIdx.x;
  const int tid  = threadIdx.x;
  const int d    = tid & 127;         // dot id within block (gate,j)
  const int half = tid >> 7;          // k-range half
  const int g    = d >> 5;
  const int jl   = d & 31;
  const int j    = b * JPB + jl;
  const float* U = (g == 0) ? Uf : (g == 1) ? Ui : (g == 2) ? Uc : Uo;
  if (tid < JPB) sc[tid] = 0.f;

  for (int t = 0; t < SEQ; ++t) {
    const float* hin = hbuf + (t & 1) * DIMS;
    for (int k = tid; k < DIMS; k += 256) sh[k] = hin[k];
    __syncthreads();

    float acc = 0.f;
    const int k0 = half * (DIMS / 2);
#pragma unroll 8
    for (int k = 0; k < DIMS / 2; ++k)
      acc = fmaf(sh[k0 + k], U[(size_t)(k0 + k) * DIMS + j], acc);
    part[tid] = acc;
    __syncthreads();

    if (tid < 128) {
      const float v = XG[(size_t)t * GDIM + g * DIMS + j] + part[tid] + part[tid + 128];
      sg[g][jl] = (g == 2) ? tanhf(v) : 1.f / (1.f + __expf(-v));
    }
    __syncthreads();

    if (tid < JPB) {
      const float cn = sg[0][tid] * sc[tid] + sg[1][tid] * sg[2][tid];
      sc[tid] = cn;
      const float hn = sg[3][tid] * tanhf(cn);
      const int jj = b * JPB + tid;
      H[(size_t)t * DIMS + jj] = hn;
      hbuf[((t + 1) & 1) * DIMS + jj] = hn;
    }

    __threadfence();
    __syncthreads();
    if (tid == 0) {
      __hip_atomic_fetch_add(barcnt, 1u, __ATOMIC_ACQ_REL, __HIP_MEMORY_SCOPE_AGENT);
      const unsigned target = (unsigned)(t + 1) * NWG_SCAN;
      while (__hip_atomic_load(barcnt, __ATOMIC_ACQUIRE, __HIP_MEMORY_SCOPE_AGENT) < target)
        __builtin_amdgcn_s_sleep(2);
    }
    __syncthreads();
    __threadfence();
  }
}

// ---------------- Kernel 3: logits = H @ V + by (packed B, 4 row tiles/wave) ---
__global__ __launch_bounds__(256) void logits_kernel(
    const float* __restrict__ H, const __bf16* __restrict__ PB,
    const float* __restrict__ by, float* __restrict__ out) {
  __shared__ __bf16 sA[ROWS_PER_BLK * DIMS];     // 64 KB
  const int row0 = blockIdx.x * ROWS_PER_BLK;
  const int tid  = threadIdx.x;
  const float4* src = (const float4*)(H + (size_t)row0 * DIMS);  // contiguous slab
  for (int e4 = tid; e4 < ROWS_PER_BLK * DIMS / 4; e4 += 256) {
    const float4 f = src[e4];
    v4bf o; o[0] = (__bf16)f.x; o[1] = (__bf16)f.y; o[2] = (__bf16)f.z; o[3] = (__bf16)f.w;
    ((v4bf*)sA)[e4] = o;
  }
  __syncthreads();

  const int wave = tid >> 5, lane = tid & 31;
  const int c = blockIdx.y * 8 + wave;
  if (c >= NTILE) return;                        // no barriers past this point

  const v16bf* pb = (const v16bf*)PB + (size_t)c * FRAG_PER_TILE + lane;
  v8f acc0 = {}, acc1 = {}, acc2 = {}, acc3 = {};
#pragma unroll 4
  for (int kc = 0; kc < 16; ++kc) {              // K chunks of 32
    __builtin_prefetch((const void*)(pb + (kc + 2) * 32), 0, 1);  // B stream ahead
    const v16bf b = pb[kc * 32];                 // one coalesced 32B load/lane
    v16bf a0 = load_a_lds(sA + 0 * 16 * DIMS, lane, kc * 32);
    v16bf a1 = load_a_lds(sA + 1 * 16 * DIMS, lane, kc * 32);
    v16bf a2 = load_a_lds(sA + 2 * 16 * DIMS, lane, kc * 32);
    v16bf a3 = load_a_lds(sA + 3 * 16 * DIMS, lane, kc * 32);
    acc0 = __builtin_amdgcn_wmma_f32_16x16x32_bf16(false, a0, false, b, (short)0, acc0, false, false);
    acc1 = __builtin_amdgcn_wmma_f32_16x16x32_bf16(false, a1, false, b, (short)0, acc1, false, false);
    acc2 = __builtin_amdgcn_wmma_f32_16x16x32_bf16(false, a2, false, b, (short)0, acc2, false, false);
    acc3 = __builtin_amdgcn_wmma_f32_16x16x32_bf16(false, a3, false, b, (short)0, acc3, false, false);
  }

  const int n = c * 16 + (lane & 15);
  if (n < VOCAB) {
    const float bv = by[n];
    const int mb = (lane >> 4) << 3;
    v8f accs[4] = {acc0, acc1, acc2, acc3};
#pragma unroll
    for (int rt = 0; rt < 4; ++rt)
#pragma unroll
      for (int r = 0; r < 8; ++r)
        out[(size_t)(row0 + rt * 16 + mb + r) * VOCAB + n] = accs[rt][r] + bv;
  }
}

// ---------------- Kernel 4: row-wise log_softmax in place ----------------
__global__ __launch_bounds__(256) void logsoftmax_kernel(float* __restrict__ out) {
  float* row = out + (size_t)blockIdx.x * VOCAB;
  __shared__ float red[256];
  const int tid = threadIdx.x;
  float m = -3.402823466e38f;
  for (int v = tid; v < VOCAB; v += 256) m = fmaxf(m, row[v]);
  red[tid] = m; __syncthreads();
  for (int s = 128; s > 0; s >>= 1) {
    if (tid < s) red[tid] = fmaxf(red[tid], red[tid + s]);
    __syncthreads();
  }
  m = red[0]; __syncthreads();
  float sum = 0.f;
  for (int v = tid; v < VOCAB; v += 256) sum += __expf(row[v] - m);
  red[tid] = sum; __syncthreads();
  for (int s = 128; s > 0; s >>= 1) {
    if (tid < s) red[tid] += red[tid + s];
    __syncthreads();
  }
  const float lse = m + __logf(red[0]);
  for (int v = tid; v < VOCAB; v += 256) row[v] = row[v] - lse;
}

// ---------------- host-side launch ----------------
extern "C" void kernel_launch(void* const* d_in, const int* in_sizes, int n_in,
                              void* d_out, int out_size, void* d_ws, size_t ws_size,
                              hipStream_t stream) {
  const int*   idx = (const int*)  d_in[0];
  const float* emb = (const float*)d_in[1];
  const float* Wf  = (const float*)d_in[2];
  const float* Wi  = (const float*)d_in[3];
  const float* Wc  = (const float*)d_in[4];
  const float* Wo  = (const float*)d_in[5];
  const float* Uf  = (const float*)d_in[6];
  const float* Ui  = (const float*)d_in[7];
  const float* Uc  = (const float*)d_in[8];
  const float* Uo  = (const float*)d_in[9];
  const float* bf  = (const float*)d_in[10];
  const float* bi  = (const float*)d_in[11];
  const float* bc  = (const float*)d_in[12];
  const float* bo  = (const float*)d_in[13];
  const float* V   = (const float*)d_in[14];
  const float* by  = (const float*)d_in[15];
  float* out = (float*)d_out;

  // workspace layout (4KB-aligned regions)
  const size_t TILE_BYTES = (size_t)FRAG_PER_TILE * 16 * sizeof(__bf16);  // 16 KB/tile
  char* ws = (char*)d_ws;
  float*    XG     = (float*)ws;                                          // 16 MB
  float*    H      = (float*)(ws + (size_t)SEQ * GDIM * 4);               //  4 MB
  float*    hbuf   = (float*)(ws + (size_t)SEQ * GDIM * 4 + (size_t)SEQ * DIMS * 4);
  unsigned* barcnt = (unsigned*)(hbuf + 2 * DIMS);
  __bf16*   PB     = (__bf16*)(ws + (size_t)SEQ * GDIM * 4 + (size_t)SEQ * DIMS * 4 + 4096);
  __bf16*   PW     = (__bf16*)((char*)PB + (size_t)NTILE * TILE_BYTES);   //  2 MB

  // pack weights to bf16 fragment order (one-time per launch)
  packB_kernel<<<NTILE, 512, 0, stream>>>(V, VOCAB, VOCAB, PB);
  packB_kernel<<<WTILE / 4, 512, 0, stream>>>(Wf, DIMS, DIMS, PW + 0 * (WTILE / 4) * TILE_BYTES / 2);
  packB_kernel<<<WTILE / 4, 512, 0, stream>>>(Wi, DIMS, DIMS, PW + 1 * (WTILE / 4) * TILE_BYTES / 2);
  packB_kernel<<<WTILE / 4, 512, 0, stream>>>(Wc, DIMS, DIMS, PW + 2 * (WTILE / 4) * TILE_BYTES / 2);
  packB_kernel<<<WTILE / 4, 512, 0, stream>>>(Wo, DIMS, DIMS, PW + 3 * (WTILE / 4) * TILE_BYTES / 2);
  scan_init_kernel<<<(2 * DIMS + 255) / 256, 256, 0, stream>>>(hbuf, barcnt);

  pregate_kernel<<<dim3(SEQ / ROWS_PER_BLK, WTILE / 8), 256, 0, stream>>>(
      idx, emb, PW, bf, bi, bc, bo, XG);
  lstm_scan_multi<<<NWG_SCAN, 256, 0, stream>>>(XG, Uf, Ui, Uc, Uo, H, hbuf, barcnt);
  logits_kernel<<<dim3(SEQ / ROWS_PER_BLK, (NTILE + 7) / 8), 256, 0, stream>>>(H, PB, by, out);
  logsoftmax_kernel<<<SEQ, 256, 0, stream>>>(out);
}
